// MultiHeadAttention_76570676953675
// MI455X (gfx1250) — compile-verified, matrix operand order checked by reading, not run on hardware
//
#include <hip/hip_runtime.h>
#include <hip/hip_bf16.h>
#include <stdint.h>

typedef __bf16 bf16_t;
typedef __attribute__((ext_vector_type(8)))  __bf16 v8bf;
typedef __attribute__((ext_vector_type(16))) __bf16 v16bf;
typedef __attribute__((ext_vector_type(8)))  float  v8f;

#define DIM   1024
#define SEQ   2048
#define NH    16
#define HD    64
#define BATCH 2
#define MTOT  (BATCH * SEQ)   // 4096 tokens

// ---------------- helpers ----------------

__device__ __forceinline__ v8f zero8() {
  v8f z;
#pragma unroll
  for (int i = 0; i < 8; ++i) z[i] = 0.0f;
  return z;
}

// Low 32 bits of a generic pointer to __shared__ = LDS byte offset (aperture mapping).
__device__ __forceinline__ unsigned lds_off(const void* p) {
  return (unsigned)(unsigned long long)(uintptr_t)p;
}

// CDNA5 async global->LDS copy (per-lane 16B), tracked by ASYNCcnt.
__device__ __forceinline__ void async_ld_b128(unsigned lds, const void* g) {
  asm volatile("global_load_async_to_lds_b128 %0, %1, off"
               :: "v"(lds), "v"((unsigned long long)(uintptr_t)g)
               : "memory");
}

template <int N>
__device__ __forceinline__ void wait_asynccnt() {
  asm volatile("s_wait_asynccnt %0" :: "i"(N) : "memory");
}

// Load a 16-element bf16 WMMA fragment: two contiguous 16B chunks at p and p+16
// (ISA 16-bit A/B layout: k = half*8 + e  and  k = 16 + half*8 + e).
__device__ __forceinline__ v16bf ldfrag(const bf16_t* p) {
  v8bf lo = *(const v8bf*)(p);
  v8bf hi = *(const v8bf*)(p + 16);
  return __builtin_shufflevector(lo, hi, 0, 1, 2, 3, 4, 5, 6, 7,
                                 8, 9, 10, 11, 12, 13, 14, 15);
}

__device__ __forceinline__ v8f wmma_bf16(v16bf a, v16bf b, v8f c) {
  return __builtin_amdgcn_wmma_f32_16x16x32_bf16(false, a, false, b,
                                                 (short)0, c, false, false);
}

// ---------------- conversion kernels ----------------

__global__ void cvt_bf16_kernel(const float* __restrict__ in,
                                bf16_t* __restrict__ out, int n) {
  int i = blockIdx.x * blockDim.x + threadIdx.x;
  if (i < n) out[i] = (bf16_t)in[i];
}

// out[c*R + r] = (bf16) in[r*C + c]  (weights as [N][K] for B-fragments)
__global__ void cvt_transpose_kernel(const float* __restrict__ in,
                                     bf16_t* __restrict__ out, int R, int C) {
  int idx = blockIdx.x * blockDim.x + threadIdx.x;
  if (idx >= R * C) return;
  int r = idx / C, c = idx % C;
  out[(size_t)c * R + r] = (bf16_t)in[idx];
}

// ---------------- bf16 GEMM: C[M,N] = A[M,K] * Bt[N,K]^T + bias ----------------
// MODE 0: scatter bf16 into qkv buffer: Q,K as [B][H][S][D]; V as [B][H][D][S].
// MODE 1: f32 output + bias, row-major [M,N].

#define ASTR 40           // padded LDS row strides (40*2 = 80B keeps 16B alignment)
#define BSTR 40
#define ABUF (64 * ASTR)
#define BBUF (256 * BSTR)

template <int MODE>
__global__ __launch_bounds__(256) void gemm_bf16_kernel(
    const bf16_t* __restrict__ A, const bf16_t* __restrict__ Bt,
    const float* __restrict__ bias, void* __restrict__ outp,
    int M, int N, int K) {
  __shared__ bf16_t Aw[2 * ABUF];   // double-buffered 64x32 A tiles
  __shared__ bf16_t Bw[2 * BBUF];   // double-buffered 256x32 B tiles

  const int tid  = threadIdx.x;
  const int wave = tid >> 5, lane = tid & 31;
  const int lm   = lane & 15, half = lane >> 4;
  const int wm   = wave >> 2, wn = wave & 3;     // 2 x 4 wave grid
  const int n0   = blockIdx.x * 256;
  const int m0   = blockIdx.y * 64;

  v8f acc[2][4];
#pragma unroll
  for (int mt = 0; mt < 2; ++mt)
#pragma unroll
    for (int nt = 0; nt < 4; ++nt) acc[mt][nt] = zero8();

  const unsigned awb = lds_off(Aw), bwb = lds_off(Bw);
  const int arow = tid >> 2, acol = (tid & 3) * 8;

  auto issue = [&](int buf, int kk) {
    // A tile: 1 async b128 per lane
    async_ld_b128(awb + (unsigned)(buf * ABUF + arow * ASTR + acol) * 2,
                  A + (size_t)(m0 + arow) * K + kk + acol);
    // B tile: 4 async b128 per lane (row n = tid)
#pragma unroll
    for (int i = 0; i < 4; ++i)
      async_ld_b128(bwb + (unsigned)(buf * BBUF + tid * BSTR + i * 8) * 2,
                    Bt + (size_t)(n0 + tid) * K + kk + i * 8);
  };

  issue(0, 0);
  for (int kk = 0; kk < K; kk += 32) {
    const int cur = (kk >> 5) & 1;
    if (kk + 32 < K) {
      issue(cur ^ 1, kk + 32);   // prefetch next tile into other buffer
      wait_asynccnt<5>();        // 5 oldest (current buffer) retired
    } else {
      wait_asynccnt<0>();
    }
    __syncthreads();

    const bf16_t* Ac = &Aw[cur * ABUF];
    const bf16_t* Bc = &Bw[cur * BBUF];
    v16bf af[2], bfg[4];
#pragma unroll
    for (int mt = 0; mt < 2; ++mt)
      af[mt] = ldfrag(&Ac[(wm * 32 + mt * 16 + lm) * ASTR + half * 8]);
#pragma unroll
    for (int nt = 0; nt < 4; ++nt)
      bfg[nt] = ldfrag(&Bc[(wn * 64 + nt * 16 + lm) * BSTR + half * 8]);
#pragma unroll
    for (int mt = 0; mt < 2; ++mt) {
#pragma unroll
      for (int nt = 0; nt < 4; ++nt)
        acc[mt][nt] = wmma_bf16(af[mt], bfg[nt], acc[mt][nt]);
    }
    __syncthreads();
  }

  // epilogue
#pragma unroll
  for (int mt = 0; mt < 2; ++mt) {
#pragma unroll
    for (int nt = 0; nt < 4; ++nt) {
#pragma unroll
      for (int r = 0; r < 8; ++r) {
        int gm = m0 + wm * 32 + mt * 16 + r + 8 * half;   // C/D: m = r + 8*half
        int gn = n0 + wn * 64 + nt * 16 + lm;             // C/D: n = lane%16
        float v = acc[mt][nt][r] + bias[gn];
        if (MODE == 0) {
          int part = gn >> 10, rem = gn & 1023;
          int hh = rem >> 6, dd = rem & 63;
          int bb = gm >> 11, ss = gm & 2047;
          size_t base = (size_t)part * ((size_t)BATCH * NH * SEQ * HD);
          size_t idx;
          if (part == 2)  // V stored head-transposed [b][h][d][s]
            idx = base + (((size_t)bb * NH + hh) * HD + dd) * SEQ + ss;
          else            // Q, K stored [b][h][s][d]
            idx = base + (((size_t)bb * NH + hh) * SEQ + ss) * HD + dd;
          ((bf16_t*)outp)[idx] = (bf16_t)v;
        } else {
          ((float*)outp)[(size_t)gm * N + gn] = v;
        }
      }
    }
  }
}

// ---------------- flash attention ----------------
// grid: (SEQ/128, BATCH*NH); block 256 threads = 8 waves; wave owns 16 q rows.
// qkv: Q,K [B][H][S][D]; V [B][H][D][S] (bf16).  aout: [B*S][DIM] bf16.

#define KVBUF (64 * HD)   // 4096 elements per K or V tile

__global__ __launch_bounds__(256) void attn_kernel(const bf16_t* __restrict__ qkv,
                                                   bf16_t* __restrict__ aout) {
  __shared__ bf16_t Kw[2 * KVBUF];      // [key][d]    16 KB (double buffered)
  __shared__ bf16_t Vt[2 * KVBUF];      // [d][key]    16 KB (double buffered)
  __shared__ bf16_t Pw[8 * 16 * 64];    // per-wave P  16 KB

  const int tid  = threadIdx.x;
  const int wave = tid >> 5, lane = tid & 31;
  const int lm   = lane & 15, half = lane >> 4;
  const int bh   = blockIdx.y;
  const int b    = bh >> 4, h = bh & 15;

  const size_t partsz = (size_t)BATCH * NH * SEQ * HD;
  const bf16_t* Qb = qkv + ((size_t)b * NH + h) * SEQ * HD;
  const bf16_t* Kb = qkv + partsz + ((size_t)b * NH + h) * SEQ * HD;
  const bf16_t* Vb = qkv + 2 * partsz + ((size_t)b * NH + h) * HD * SEQ;  // [d][s]

  const int q0 = blockIdx.x * 128 + wave * 16;

  // Q fragments (A operand), loaded once: k-dim = 64 -> 2 chunks of 32
  v16bf aq[2];
#pragma unroll
  for (int kh = 0; kh < 2; ++kh)
    aq[kh] = ldfrag(Qb + (size_t)(q0 + lm) * HD + kh * 32 + half * 8);

  v8f o[4];
#pragma unroll
  for (int t = 0; t < 4; ++t) o[t] = zero8();
  float m_row[8], l_row[8];
#pragma unroll
  for (int r = 0; r < 8; ++r) { m_row[r] = -1e30f; l_row[r] = 0.0f; }

  bf16_t* Pww = &Pw[wave * 16 * 64];
  const unsigned kwb = lds_off(Kw), vtb = lds_off(Vt);
  const int ldr = tid >> 2, ldc = (tid & 3) * 16;
  const float scale = 0.125f;   // 1/sqrt(64)

  auto issue_kv = [&](int buf, int j0) {
    // K block [64 keys][64 d] -> Kw[key][d]
    const bf16_t* kg = Kb + (size_t)(j0 + ldr) * HD + ldc;
    unsigned kl = kwb + (unsigned)(buf * KVBUF + ldr * HD + ldc) * 2;
    async_ld_b128(kl, kg);
    async_ld_b128(kl + 16, kg + 8);
    // V block (global already [d][s]) -> Vt[d][key]
    const bf16_t* vg = Vb + (size_t)ldr * SEQ + j0 + ldc;
    unsigned vl = vtb + (unsigned)(buf * KVBUF + ldr * 64 + ldc) * 2;
    async_ld_b128(vl, vg);
    async_ld_b128(vl + 16, vg + 8);
  };

  issue_kv(0, 0);
  for (int jb = 0; jb < SEQ / 64; ++jb) {
    const int cur = jb & 1;
    if (jb + 1 < SEQ / 64) {
      issue_kv(cur ^ 1, (jb + 1) * 64);
      wait_asynccnt<4>();
    } else {
      wait_asynccnt<0>();
    }
    __syncthreads();

    const bf16_t* Kc = &Kw[cur * KVBUF];
    const bf16_t* Vc = &Vt[cur * KVBUF];

    // ---- scores S = Q * K^T  (16 x 64 per wave, 8 WMMAs) ----
    v8f s[4];
#pragma unroll
    for (int t = 0; t < 4; ++t) s[t] = zero8();
#pragma unroll
    for (int kh = 0; kh < 2; ++kh) {
#pragma unroll
      for (int t = 0; t < 4; ++t) {
        v16bf bk = ldfrag(&Kc[(t * 16 + lm) * HD + kh * 32 + half * 8]);
        s[t] = wmma_bf16(aq[kh], bk, s[t]);
      }
    }
#pragma unroll
    for (int t = 0; t < 4; ++t)
#pragma unroll
      for (int r = 0; r < 8; ++r) s[t][r] *= scale;

    // ---- online softmax per row (row = r + 8*half; 16 lanes hold a row) ----
#pragma unroll
    for (int r = 0; r < 8; ++r) {
      float mx = fmaxf(fmaxf(s[0][r], s[1][r]), fmaxf(s[2][r], s[3][r]));
#pragma unroll
      for (int d = 1; d < 16; d <<= 1) mx = fmaxf(mx, __shfl_xor(mx, d));
      float mn   = fmaxf(m_row[r], mx);
      float corr = __expf(m_row[r] - mn);
      m_row[r]   = mn;
      float lb = 0.0f;
#pragma unroll
      for (int t = 0; t < 4; ++t) {
        float p = __expf(s[t][r] - mn);
        s[t][r] = p;
        lb += p;
      }
#pragma unroll
      for (int d = 1; d < 16; d <<= 1) lb += __shfl_xor(lb, d);
      l_row[r] = l_row[r] * corr + lb;
#pragma unroll
      for (int t = 0; t < 4; ++t) o[t][r] *= corr;
    }

    // ---- write P (bf16) to wave-private LDS, layout [qrow][key] ----
#pragma unroll
    for (int t = 0; t < 4; ++t)
#pragma unroll
      for (int r = 0; r < 8; ++r)
        Pww[(r + 8 * half) * 64 + t * 16 + lm] = (bf16_t)s[t][r];

    // ---- O += P * V  (8 WMMAs; B-fragments from [d][key] V) ----
#pragma unroll
    for (int kh = 0; kh < 2; ++kh) {
      v16bf ap = ldfrag(&Pww[lm * 64 + kh * 32 + half * 8]);
#pragma unroll
      for (int t = 0; t < 4; ++t) {
        v16bf bv = ldfrag(&Vc[(t * 16 + lm) * 64 + kh * 32 + half * 8]);
        o[t] = wmma_bf16(ap, bv, o[t]);
      }
    }
    __syncthreads();
  }

  // ---- finalize: divide by l, store bf16 into [B*S][DIM] ----
#pragma unroll
  for (int r = 0; r < 8; ++r) {
    float inv = 1.0f / l_row[r];
    int srow = q0 + r + 8 * half;
#pragma unroll
    for (int t = 0; t < 4; ++t) {
      int col = h * HD + t * 16 + lm;
      aout[((size_t)b * SEQ + srow) * DIM + col] = (bf16_t)(o[t][r] * inv);
    }
  }
}

// ---------------- launch ----------------

extern "C" void kernel_launch(void* const* d_in, const int* in_sizes, int n_in,
                              void* d_out, int out_size, void* d_ws, size_t ws_size,
                              hipStream_t stream) {
  const float* x      = (const float*)d_in[0];
  const float* qkv_w  = (const float*)d_in[1];
  const float* qkv_b  = (const float*)d_in[2];
  const float* proj_w = (const float*)d_in[3];
  const float* proj_b = (const float*)d_in[4];

  const size_t MB = 1u << 20;
  char* ws = (char*)d_ws;
  bf16_t* xb     = (bf16_t*)(ws);             //  8 MB: x as bf16 [4096][1024]
  bf16_t* wqkvT  = (bf16_t*)(ws + 8 * MB);    //  6 MB: qkv_w^T [3072][1024]
  bf16_t* wprojT = (bf16_t*)(ws + 14 * MB);   //  2 MB: proj_w^T [1024][1024]
  bf16_t* qkvbuf = (bf16_t*)(ws + 16 * MB);   // 24 MB: Q,K [B][H][S][D]; V [B][H][D][S]
  bf16_t* aout   = (bf16_t*)(ws + 40 * MB);   //  8 MB: attn out [4096][1024]

  // convert / transpose to bf16
  const int nx = MTOT * DIM;
  cvt_bf16_kernel<<<nx / 256, 256, 0, stream>>>(x, xb, nx);
  const int nq = DIM * 3 * DIM;
  cvt_transpose_kernel<<<(nq + 255) / 256, 256, 0, stream>>>(qkv_w, wqkvT, DIM, 3 * DIM);
  const int np = DIM * DIM;
  cvt_transpose_kernel<<<(np + 255) / 256, 256, 0, stream>>>(proj_w, wprojT, DIM, DIM);

  // QKV GEMM: [4096,1024] x [1024,3072] -> scattered bf16 Q/K/V
  gemm_bf16_kernel<0><<<dim3(3 * DIM / 256, MTOT / 64), 256, 0, stream>>>(
      xb, wqkvT, qkv_b, qkvbuf, MTOT, 3 * DIM, DIM);

  // attention
  attn_kernel<<<dim3(SEQ / 128, BATCH * NH), 256, 0, stream>>>(qkvbuf, aout);

  // projection GEMM: [4096,1024] x [1024,1024] + bias -> f32 out
  gemm_bf16_kernel<1><<<dim3(DIM / 256, MTOT / 64), 256, 0, stream>>>(
      aout, wprojT, proj_b, d_out, MTOT, DIM, DIM);
}
